// SLSTMCell_58935541235873
// MI455X (gfx1250) — compile-verified, matrix operand order checked by reading, not run on hardware
//
#include <hip/hip_runtime.h>
#include <hip/hip_bf16.h>
#include <math.h>

typedef __attribute__((ext_vector_type(16))) __bf16 bf16x16;
typedef __attribute__((ext_vector_type(8)))  float  v8f;
typedef __attribute__((ext_vector_type(4)))  int    v4i;
typedef unsigned short u16;

#define LSEQ 256
#define BB   32
#define DHH  512
#define MTOT 8192            // LSEQ*BB
#define HSZ  4210688         // 257*32*512
#define GOFF 8192            // m-index of global node row block (256*32)

#if __has_builtin(__builtin_amdgcn_global_load_async_to_lds_b128) && \
    __has_builtin(__builtin_amdgcn_s_wait_asynccnt)
#define HAVE_ASYNC 1
#else
#define HAVE_ASYNC 0
#endif

// ---- helpers ---------------------------------------------------------------

__device__ __forceinline__ unsigned bfpack(float lo, float hi) {
  // round-to-nearest-even f32 -> bf16, packed pair (one-time conversion pass only)
  unsigned a = __builtin_bit_cast(unsigned, lo);
  unsigned b = __builtin_bit_cast(unsigned, hi);
  a = (a + 0x7FFFu + ((a >> 16) & 1u)) >> 16;
  b = (b + 0x7FFFu + ((b >> 16) & 1u)) & 0xFFFF0000u;
  return a | b;
}

__device__ __forceinline__ bf16x16 fragLoadL(const unsigned* s, int dw) {  // LDS, dword offset
  struct S { uint4 a, b; } t;
  t.a = *(const uint4*)(s + dw);
  t.b = *(const uint4*)(s + dw + 4);
  return __builtin_bit_cast(bf16x16, t);
}

__device__ __forceinline__ bf16x16 fragLoadG(const u16* p) {               // global, 16 bf16
  struct S { uint4 a, b; } t;
  t.a = *(const uint4*)(p);
  t.b = *(const uint4*)(p + 8);
  return __builtin_bit_cast(bf16x16, t);
}

// stage 16 bytes global -> LDS (async-to-LDS on CDNA5 if the builtin exists)
__device__ __forceinline__ void stage16(const u16* g, unsigned* l) {
#if HAVE_ASYNC
  __builtin_amdgcn_global_load_async_to_lds_b128(
      (__attribute__((address_space(1))) v4i*)g,
      (__attribute__((address_space(3))) v4i*)l, 0, 0);
#else
  *(uint4*)l = *(const uint4*)g;
#endif
}

__device__ __forceinline__ void stageFence() {
#if HAVE_ASYNC
  __builtin_amdgcn_s_wait_asynccnt(0);
#endif
  __syncthreads();
}

__device__ __forceinline__ float sigf(float x) { return 1.f / (1.f + __expf(-x)); }

// ---- K0a: contiguous f32 -> bf16 (h0 word rows, src, s_uf) -----------------

__global__ void k_cvt(const float* __restrict__ in, u16* __restrict__ out, int ntask) {
  int t = blockIdx.x * 256 + threadIdx.x;        // 8 elems per task
  if (t >= ntask) return;
  const float4* s = (const float4*)(in + t * 8);
  float4 v0 = s[0], v1 = s[1];
  *(uint4*)(out + t * 8) = make_uint4(bfpack(v0.x, v0.y), bfpack(v0.z, v0.w),
                                      bfpack(v1.x, v1.y), bfpack(v1.z, v1.w));
}

// ---- K0b: pack 7 gate weights into wbf[(g*512+h)*2048 + k] bf16 ------------

struct WPackArgs { const float* ww[7]; const float* wu[7]; u16* wbf; };

__global__ void k_wpack(WPackArgs P) {
  int t = blockIdx.x * 256 + threadIdx.x;        // 917504 tasks, 8 elems each
  int g   = t >> 17;                             // 131072 tasks / gate
  int rem = t & 131071;
  int h   = rem >> 8;                            // 256 chunks / row
  int k   = (rem & 255) * 8;
  const float* s = (k < 1536) ? (P.ww[g] + h * 1536 + k)
                              : (P.wu[g] + h * DHH + (k - 1536));
  float4 v0 = ((const float4*)s)[0], v1 = ((const float4*)s)[1];
  *(uint4*)(P.wbf + ((g * DHH + h) * 2048 + k)) =
      make_uint4(bfpack(v0.x, v0.y), bfpack(v0.z, v0.w),
                 bfpack(v1.x, v1.y), bfpack(v1.z, v1.w));
}

// ---- K1: h_hat[b,h] = mean_n h0[n,b,h] ------------------------------------

__global__ void k_hhat(const float* __restrict__ h0, float* __restrict__ hhat) {
  int idx = blockIdx.x * 256 + threadIdx.x;      // 16384
  int b = idx >> 9, h = idx & 511;
  const float* p = h0 + b * DHH + h;
  float s = 0.f;
  for (int n = 0; n < LSEQ; ++n) s += p[n * (BB * DHH)];
  hhat[idx] = s * (1.f / 256.f);
}

// ---- K2: per-batch small GEMMs (biases, fg, og_s, fhg) ---------------------

struct SmallArgs {
  const float* h0;
  const float* hhat;
  const float* wv[7];
  const float* wb[7];
  const float* s_wg; const float* s_ug; const float* s_bg;
  const float* s_wf; const float* s_bf;
  const float* s_wo; const float* s_uo; const float* s_bo;
  float* bias; float* fg; float* ogs; float* fhg;
};

__device__ __forceinline__ float dot512(const float* x, const float* y) {
  const float4* X = (const float4*)x; const float4* Y = (const float4*)y;
  float s = 0.f;
  #pragma unroll 8
  for (int i = 0; i < 128; ++i) {
    float4 a = X[i], b = Y[i];
    s += a.x*b.x + a.y*b.y + a.z*b.z + a.w*b.w;
  }
  return s;
}

__global__ void k_small(SmallArgs A) {
  int gid = blockIdx.x * 256 + threadIdx.x;      // 163840 total
  int cls = gid >> 14;
  int idx = gid & 16383;
  int b = idx >> 9, h = idx & 511;
  const float* hgt = A.h0 + (GOFF + b) * DHH;    // prev_h_gt row
  if (cls < 7) {
    A.bias[cls * 16384 + idx] = dot512(hgt, A.wv[cls] + h * DHH) + A.wb[cls][h];
  } else if (cls == 7) {
    float v = dot512(hgt, A.s_wg + h*DHH) + dot512(A.hhat + b*DHH, A.s_ug + h*DHH) + A.s_bg[h];
    A.fg[idx] = sigf(v);
  } else if (cls == 8) {
    float v = dot512(hgt, A.s_wo + h*DHH) + dot512(A.hhat + b*DHH, A.s_uo + h*DHH) + A.s_bo[h];
    A.ogs[idx] = sigf(v);
  } else {
    A.fhg[idx] = dot512(hgt, A.s_wf + h*DHH) + A.s_bf[h];
  }
}

// ---- K3: fi = sigmoid(h_prev @ s_uf^T + fhg)   M=8192 K=512 N=512 ---------
// A (hbf rows) staged in LDS (shared by waves), B (ufbf) direct global->VGPR.

__global__ void __launch_bounds__(256) k_fi(const u16* __restrict__ hbf,
                                            const u16* __restrict__ ufbf,
                                            const float* __restrict__ fhg,
                                            float* __restrict__ fi) {
  __shared__ __align__(16) unsigned shm[2560];   // A: 128 rows x 32 bf16 (stride 40)
  int tid  = threadIdx.x;
  int lane = tid & 31, wave = tid >> 5;
  int mBase = blockIdx.x * 128, hBase = blockIdx.y * 64;
  int wm = wave >> 1, wn = wave & 1;             // wave -> 32x32 subtile
  int mrow = lane & 15, khalf = lane >> 4;

  v8f acc[2][2] = {};
  for (int kt = 0; kt < 16; ++kt) {
    int kb = kt * 32;
    #pragma unroll
    for (int t = 0; t < 2; ++t) {                // 512 16B-chunk tasks
      int task = tid + t * 256;
      int row = task >> 2, chunk = task & 3;
      stage16(hbf + (mBase + row) * DHH + kb + chunk * 8, &shm[row * 20 + chunk * 4]);
    }
    stageFence();
    bf16x16 af[2], bfr[2];
    #pragma unroll
    for (int i = 0; i < 2; ++i) af[i] = fragLoadL(shm, (wm*32 + i*16 + mrow) * 20 + khalf * 8);
    #pragma unroll
    for (int i = 0; i < 2; ++i) {
      int h = hBase + wn*32 + i*16 + (lane & 15);
      bfr[i] = fragLoadG(ufbf + h * DHH + kb + khalf * 16);
    }
    #pragma unroll
    for (int i = 0; i < 2; ++i)
      #pragma unroll
      for (int j = 0; j < 2; ++j)
        acc[i][j] = __builtin_amdgcn_wmma_f32_16x16x32_bf16(false, af[i], false, bfr[j],
                                                            (short)0, acc[i][j], false, false);
    __syncthreads();
  }
  #pragma unroll
  for (int i = 0; i < 2; ++i)
    for (int j = 0; j < 2; ++j)
      #pragma unroll
      for (int r = 0; r < 8; ++r) {
        int mg = mBase + wm*32 + i*16 + r + khalf*8;   // C layout: M = vgpr + 8*(lane>=16)
        int hg = hBase + wn*32 + j*16 + (lane & 15);
        fi[mg * DHH + hg] = sigf(acc[i][j][r] + fhg[(mg & 31) * DHH + hg]);
      }
}

// ---- K4: softmax over n + global node update -------------------------------

__global__ void k_global(const float* __restrict__ fi, const float* __restrict__ c0,
                         const float* __restrict__ fg, const float* __restrict__ ogs,
                         float* __restrict__ out) {
  int idx = blockIdx.x * 256 + threadIdx.x;      // 16384
  int b = idx >> 9, h = idx & 511;
  const float* fp = fi + b * DHH + h;
  const float* cp = c0 + b * DHH + h;
  float sum = 0.f;
  for (int n = 0; n < LSEQ; ++n) sum += __expf(fp[n * 16384]);  // fi in (0,1): no max needed
  float acc = 0.f;
  for (int n = 0; n < LSEQ; ++n) acc += __expf(fp[n * 16384]) * cp[n * 16384];
  float cg = fg[idx] * c0[(GOFF + b) * DHH + h] + acc / sum;
  float hg = ogs[idx] * tanhf(cg);
  out[(GOFF + b) * DHH + h]       = hg;
  out[HSZ + (GOFF + b) * DHH + h] = cg;
}

// ---- K5: main fused GEMM (M=8192, K=2048, 7 gates x 64h per WG) ------------
// A tile (shift-windowed h_prev / src, bf16) via async LDS staging, 7-way reuse.
// B fragments loaded directly global->VGPR from the pre-packed L2-resident wbf.

struct MainArgs {
  const u16* hbf; const u16* sbf; const u16* wbf;
  const float* c0; const float* bias;
  float* out;
};

__global__ void __launch_bounds__(256) k_main(MainArgs A) {
  __shared__ __align__(16) unsigned shm[1280 + 1792];  // A 64x32 bf16 (stride 40) + 7x16x16 f32
  float* ePf = (float*)&shm[1280];
  int tid  = threadIdx.x;
  int lane = tid & 31, wave = tid >> 5;                // waves 0..6 compute gate `wave`
  int mrow = lane & 15, khalf = lane >> 4;
  int mBase = blockIdx.x * 64, hBase = blockIdx.y * 64;
  const u16* wgate = A.wbf + (size_t)wave * DHH * 2048;

  v8f acc[4][4] = {};

  for (int kt = 0; kt < 64; ++kt) {
    int kb = kt * 32;
    {                                                  // stage A: 256 16B-chunk tasks
      int row = tid >> 2, chunk = tid & 3;
      int reg = kb >> 9, kk = (kb & 511) + chunk * 8;
      int m = mBase + row;
      const u16* src = nullptr;
      bool valid = true;
      if (reg == 0)      { valid = (m >= 32);        src = A.hbf + (m - 32) * DHH + kk; }
      else if (reg == 1) {                            src = A.hbf + m * DHH + kk; }
      else if (reg == 2) { valid = (m + 32 < MTOT);  src = A.hbf + (m + 32) * DHH + kk; }
      else               {                            src = A.sbf + m * DHH + kk; }
      unsigned* dst = &shm[row * 20 + chunk * 4];
      if (valid) {
        __builtin_prefetch((const void*)(src + 32), 0, 0);   // next K slice
        stage16(src, dst);
      } else {
        *(uint4*)dst = make_uint4(0, 0, 0, 0);
      }
    }
    stageFence();
    if (wave < 7) {                                    // wave-uniform: EXEC all ones for WMMA
      bf16x16 af[4], bfr[4];
      #pragma unroll
      for (int i = 0; i < 4; ++i) af[i] = fragLoadL(shm, (i*16 + mrow) * 20 + khalf * 8);
      #pragma unroll
      for (int i = 0; i < 4; ++i) {
        int h = hBase + i*16 + (lane & 15);
        bfr[i] = fragLoadG(wgate + (size_t)h * 2048 + kb + khalf * 16);
      }
      #pragma unroll
      for (int i = 0; i < 4; ++i)
        #pragma unroll
        for (int j = 0; j < 4; ++j)
          acc[i][j] = __builtin_amdgcn_wmma_f32_16x16x32_bf16(false, af[i], false, bfr[j],
                                                              (short)0, acc[i][j], false, false);
    }
    __syncthreads();
  }

  // fused epilogue: 16 rounds of 16x16 subtiles, cross-gate softmax via LDS
  for (int r = 0; r < 16; ++r) {
    int mi = r >> 2, ni = r & 3;
    if (wave < 7) {
      #pragma unroll
      for (int j = 0; j < 8; ++j)
        ePf[wave * 256 + (j + khalf * 8) * 16 + (lane & 15)] = acc[mi][ni][j];
    }
    __syncthreads();
    {
      int m = mBase + mi * 16 + (tid >> 4);
      int h = hBase + ni * 16 + (tid & 15);
      int b = m & 31;
      float p[7];                                      // 0=i 1=l 2=r 3=f 4=s 5=o 6=u
      #pragma unroll
      for (int t = 0; t < 7; ++t)
        p[t] = ePf[t * 256 + tid] + A.bias[t * 16384 + b * DHH + h];
      float gl = p[1], gf = p[3], gr = p[2], gs = p[4], gi = p[0];
      float mx = fmaxf(fmaxf(fmaxf(gl, gf), fmaxf(gr, gs)), gi);
      float el = __expf(gl - mx), ef = __expf(gf - mx), er = __expf(gr - mx),
            es = __expf(gs - mx), ei = __expf(gi - mx);
      float inv = 1.f / (el + ef + er + es + ei);
      float u  = tanhf(p[6]);
      float og = sigf(p[5]);
      float cl  = (m >= 32)        ? A.c0[(m - 32) * DHH + h] : 0.f;
      float cc  =                    A.c0[m * DHH + h];
      float cr  = (m + 32 < MTOT)  ? A.c0[(m + 32) * DHH + h] : 0.f;
      float cgp =                    A.c0[(GOFF + b) * DHH + h];
      float cw = (el * cl + ef * cc + er * cr + es * cgp + ei * u) * inv;
      float hw = og * tanhf(cw);
      A.out[m * DHH + h]       = hw;
      A.out[HSZ + m * DHH + h] = cw;
    }
    __syncthreads();
  }
}

// ---- host ------------------------------------------------------------------

extern "C" void kernel_launch(void* const* d_in, const int* in_sizes, int n_in,
                              void* d_out, int out_size, void* d_ws, size_t ws_size,
                              hipStream_t stream) {
  const float* src = (const float*)d_in[0];
  const float* h0  = (const float*)d_in[2];
  const float* c0  = (const float*)d_in[3];
  float* W    = (float*)d_ws;
  float* hhat = W;                 // 16384 f32
  float* bias = W + 16384;         // 7*16384
  float* fg   = W + 131072;        // 16384
  float* ogs  = W + 147456;        // 16384
  float* fhg  = W + 163840;        // 16384
  float* fi   = W + 180224;        // 8192*512
  u16* US   = (u16*)(W + 4374528); // bf16 region (16B aligned)
  u16* hbf  = US;                  // 8192*512
  u16* sbf  = US + 4194304;        // 8192*512
  u16* ufbf = US + 8388608;        // 512*512
  u16* wbf  = US + 8650752;        // 7*512*2048
  float* out = (float*)d_out;

  // one-time precision conversion / weight packing (bandwidth pass)
  k_cvt<<<2048, 256, 0, stream>>>(h0,  hbf, 524288);
  k_cvt<<<2048, 256, 0, stream>>>(src, sbf, 524288);
  k_cvt<<<128,  256, 0, stream>>>((const float*)d_in[36], ufbf, 32768);
  WPackArgs wp;
  for (int g = 0; g < 7; ++g) {
    wp.ww[g] = (const float*)d_in[4 + g*4 + 0];
    wp.wu[g] = (const float*)d_in[4 + g*4 + 1];
  }
  wp.wbf = wbf;
  k_wpack<<<3584, 256, 0, stream>>>(wp);

  k_hhat<<<64, 256, 0, stream>>>(h0, hhat);

  SmallArgs sa;
  sa.h0 = h0; sa.hhat = hhat;
  for (int g = 0; g < 7; ++g) {
    sa.wv[g] = (const float*)d_in[4 + g*4 + 2];
    sa.wb[g] = (const float*)d_in[4 + g*4 + 3];
  }
  sa.s_wg = (const float*)d_in[32]; sa.s_ug = (const float*)d_in[33]; sa.s_bg = (const float*)d_in[34];
  sa.s_wf = (const float*)d_in[35]; sa.s_bf = (const float*)d_in[37];
  sa.s_wo = (const float*)d_in[38]; sa.s_uo = (const float*)d_in[39]; sa.s_bo = (const float*)d_in[40];
  sa.bias = bias; sa.fg = fg; sa.ogs = ogs; sa.fhg = fhg;
  k_small<<<640, 256, 0, stream>>>(sa);

  k_fi<<<dim3(64, 8), 256, 0, stream>>>(hbf, ufbf, fhg, fi);

  k_global<<<64, 256, 0, stream>>>(fi, c0, fg, ogs, out);

  MainArgs ma;
  ma.hbf = hbf; ma.sbf = sbf; ma.wbf = wbf;
  ma.c0 = c0; ma.bias = bias; ma.out = out;
  k_main<<<dim3(128, 8), 256, 0, stream>>>(ma);
}
